// RetNetBlock_55405078118792
// MI455X (gfx1250) — compile-verified, matrix operand order checked by reading, not run on hardware
//
#include <hip/hip_runtime.h>
#include <cstdint>

// ---------------------------------------------------------------------------
// RetNet block for MI455X (gfx1250): LN1 -> QKV (WMMA f16) -> chunked diagonal
// scan -> residual -> LN2 -> MLP (WMMA f16) -> residual.
// Memory-bound problem: all GEMM operands are f16 (halve HBM bytes), f32 acc.
// ---------------------------------------------------------------------------

typedef _Float16 half_t;
typedef __attribute__((ext_vector_type(16))) _Float16 v16h;
typedef __attribute__((ext_vector_type(8)))  float    v8f;
typedef unsigned int u32x4 __attribute__((ext_vector_type(4)));
typedef int          i32x8 __attribute__((ext_vector_type(8)));
typedef int          i32x4 __attribute__((ext_vector_type(4)));

#ifndef USE_TDM
#  if defined(__has_builtin)
#    if __has_builtin(__builtin_amdgcn_tensor_load_to_lds)
#      define USE_TDM 1
#    else
#      define USE_TDM 0
#    endif
#  else
#    define USE_TDM 0
#  endif
#endif

constexpr int Bb  = 8;
constexpr int Tt  = 8192;
constexpr int Dd  = 256;
constexpr int Hh  = 1024;
constexpr int BT  = Bb * Tt;          // 65536 tokens
constexpr int NC  = 64;               // scan chunks
constexpr int Lc  = Tt / NC;          // 128 steps per chunk
constexpr int HSTR = 264;             // padded f16 LDS row stride (breaks bank conflicts)

// ---- workspace layout (bytes) ----
constexpr size_t SZ_WDD = (size_t)Dd * Dd * 2;
constexpr size_t OFF_WQ = 0;
constexpr size_t OFF_WK = OFF_WQ + SZ_WDD;
constexpr size_t OFF_WV = OFF_WK + SZ_WDD;
constexpr size_t SZ_W1  = (size_t)Hh * Dd * 2;
constexpr size_t OFF_W1 = OFF_WV + SZ_WDD;
constexpr size_t SZ_W2  = (size_t)Dd * Hh * 2;
constexpr size_t OFF_W2 = OFF_W1 + SZ_W1;
constexpr size_t SZ_Q16 = (size_t)BT * Dd * 2;
constexpr size_t OFF_Q16 = OFF_W2 + SZ_W2;
constexpr size_t SZ_KV  = (size_t)BT * Dd * 4;
constexpr size_t OFF_KV = OFF_Q16 + SZ_Q16;
constexpr size_t SZ_CAR = (size_t)Bb * NC * Dd * 4;
constexpr size_t OFF_CAR = OFF_KV + SZ_KV;
constexpr size_t OFF_CIN = OFF_CAR + SZ_CAR;
constexpr size_t OFF_HID = OFF_CIN + SZ_CAR;   // BT*H f16 = 128 MB

// ---------------------------------------------------------------------------
// WMMA helpers (ISA 7.12.2 layouts)
// ---------------------------------------------------------------------------
__device__ __forceinline__ v8f wmma_f16(v16h a, v16h b, v8f c) {
  return __builtin_amdgcn_wmma_f32_16x16x32_f16(false, a, false, b, (short)0, c,
                                                false, false);
}

// A-matrix 16x32 f16 fragment (also used for B: rows of W == columns of B).
// lane<16: M=lane, K pairs {0,1},{2,3},{4,5},{6,7},{16,17},{18,19},{20,21},{22,23}
// lane>=16: same M, K offset +8.
__device__ __forceinline__ v16h frag_rows(const half_t* base, int ld, int row0,
                                          int lane, int kbase) {
  const int m = row0 + (lane & 15);
  const int khalf = (lane & 16) ? 8 : 0;
  const half_t* p = base + (size_t)m * ld;
  v16h f;
#pragma unroll
  for (int v = 0; v < 8; ++v) {
    const int k = kbase + khalf + ((v < 4) ? (2 * v) : (16 + 2 * (v - 4)));
    f[2 * v]     = p[k];
    f[2 * v + 1] = p[k + 1];
  }
  return f;
}

// A fragment from the TDM-padded hidden tile (5 pad dwords per 256 dwords).
__device__ __forceinline__ v16h frag_padded(const uint32_t* sh, int row0,
                                            int lane, int kbase) {
  const int m = row0 + (lane & 15);
  const int khalf = (lane & 16) ? 8 : 0;
  const half_t* hp = (const half_t*)sh;
  v16h f;
#pragma unroll
  for (int v = 0; v < 8; ++v) {
    const int k = kbase + khalf + ((v < 4) ? (2 * v) : (16 + 2 * (v - 4)));
    const int idx = m * 512 + (k >> 1);          // dword index, unpadded
    const int p = idx + 5 * (idx >> 8);          // apply TDM LDS padding
    f[2 * v]     = hp[2 * p];
    f[2 * v + 1] = hp[2 * p + 1];
  }
  return f;
}

// ---------------------------------------------------------------------------
// K0: f32 -> f16 weight conversion
// ---------------------------------------------------------------------------
__global__ __launch_bounds__(256) void k_cvt_f16(const float* __restrict__ src,
                                                 half_t* __restrict__ dst, int n) {
  int i = blockIdx.x * 256 + threadIdx.x;
  if (i < n) dst[i] = (half_t)src[i];
}

// ---------------------------------------------------------------------------
// K1: fused LN1 + QKV projection. 32 tokens/block, 8 waves, WMMA 16x16x32.
// Writes q (f16) and kv = k*v (f32).
// ---------------------------------------------------------------------------
__global__ __launch_bounds__(256) void k_ln_qkv(
    const float* __restrict__ x, const float* __restrict__ g,
    const float* __restrict__ be, const half_t* __restrict__ Wq,
    const half_t* __restrict__ Wk, const half_t* __restrict__ Wv,
    const float* __restrict__ bq, const float* __restrict__ bk,
    const float* __restrict__ bv, half_t* __restrict__ q16,
    float* __restrict__ kv) {
  __shared__ half_t hA[32 * HSTR];
  const int tok0 = blockIdx.x * 32;
  const int wave = threadIdx.x >> 5, lane = threadIdx.x & 31;

  // --- LayerNorm: wave w normalizes rows 4w..4w+3 (wave32 reductions) ---
#pragma unroll
  for (int rr = 0; rr < 4; ++rr) {
    const int row = wave * 4 + rr;
    const float* xr = x + (size_t)(tok0 + row) * Dd;
    float xv[8], s = 0.f, sq = 0.f;
#pragma unroll
    for (int i = 0; i < 8; ++i) {
      float v = xr[lane + 32 * i];
      xv[i] = v; s += v; sq += v * v;
    }
#pragma unroll
    for (int o = 16; o > 0; o >>= 1) {
      s += __shfl_xor(s, o, 32);
      sq += __shfl_xor(sq, o, 32);
    }
    const float mu = s * (1.f / Dd);
    const float rstd = rsqrtf(sq * (1.f / Dd) - mu * mu + 1e-5f);
#pragma unroll
    for (int i = 0; i < 8; ++i) {
      const int c = lane + 32 * i;
      hA[row * HSTR + c] = (half_t)((xv[i] - mu) * rstd * g[c] + be[c]);
    }
  }
  __syncthreads();

  // --- q/k/v GEMMs: each wave owns 2 N-tiles x 2 M-tiles ---
  for (int mt = 0; mt < 2; ++mt) {
    for (int ni = 0; ni < 2; ++ni) {
      const int nt = wave * 2 + ni;
      __builtin_prefetch(Wq + (size_t)(nt * 16 + (lane & 15)) * Dd, 0, 1);
      __builtin_prefetch(Wk + (size_t)(nt * 16 + (lane & 15)) * Dd, 0, 1);
      __builtin_prefetch(Wv + (size_t)(nt * 16 + (lane & 15)) * Dd, 0, 1);
      v8f aq = {}, ak = {}, av = {};
#pragma unroll
      for (int kk = 0; kk < 8; ++kk) {
        const v16h af = frag_rows(hA, HSTR, mt * 16, lane, kk * 32);
        const v16h fq = frag_rows(Wq, Dd, nt * 16, lane, kk * 32);
        const v16h fk = frag_rows(Wk, Dd, nt * 16, lane, kk * 32);
        const v16h fv = frag_rows(Wv, Dd, nt * 16, lane, kk * 32);
        aq = wmma_f16(af, fq, aq);
        ak = wmma_f16(af, fk, ak);
        av = wmma_f16(af, fv, av);
      }
      const int col = nt * 16 + (lane & 15);
      const float bqv = bq[col], bkv = bk[col], bvv = bv[col];
#pragma unroll
      for (int r = 0; r < 8; ++r) {
        const int row = tok0 + mt * 16 + r + ((lane & 16) ? 8 : 0);
        const size_t idx = (size_t)row * Dd + col;
        q16[idx] = (half_t)(aq[r] + bqv);
        kv[idx] = (ak[r] + bkv) * (av[r] + bvv);
      }
    }
  }
}

// ---------------------------------------------------------------------------
// K2: chunk-local scan (in place), emits per-chunk carry. 512 blocks.
// ---------------------------------------------------------------------------
__global__ __launch_bounds__(256) void k_scan_chunks(float* __restrict__ kv,
                                                     const float* __restrict__ decay,
                                                     float* __restrict__ carry) {
  const int b = blockIdx.x / NC, c = blockIdx.x % NC, d = threadIdx.x;
  const float dec = decay[d];
  float m = 0.f;
  const size_t base = ((size_t)b * Tt + (size_t)c * Lc) * Dd + d;
  for (int i = 0; i < Lc; ++i) {
    const size_t idx = base + (size_t)i * Dd;
    m = dec * m + kv[idx];
    kv[idx] = m;   // local scan value
  }
  carry[((size_t)b * NC + c) * Dd + d] = m;
}

// ---------------------------------------------------------------------------
// K3: serial scan across 64 chunk carries (decay^128 step). Tiny.
// ---------------------------------------------------------------------------
__global__ __launch_bounds__(256) void k_scan_carry(const float* __restrict__ carry,
                                                    float* __restrict__ cin,
                                                    const float* __restrict__ decay) {
  const int b = blockIdx.x, d = threadIdx.x;
  const float dec = decay[d];
  float dl = dec;
#pragma unroll
  for (int i = 0; i < 7; ++i) dl *= dl;   // dec^128
  float m = 0.f;
  for (int c = 0; c < NC; ++c) {
    const size_t idx = ((size_t)b * NC + c) * Dd + d;
    cin[idx] = m;
    m = dl * m + carry[idx];
  }
}

// ---------------------------------------------------------------------------
// K4: ret = q * (local + dec^(i+1)*carry_in); out = x + ret
// ---------------------------------------------------------------------------
__global__ __launch_bounds__(256) void k_ret_residual(
    const float* __restrict__ x, const half_t* __restrict__ q16,
    const float* __restrict__ kvloc, const float* __restrict__ cin,
    const float* __restrict__ decay, float* __restrict__ out) {
  const int b = blockIdx.x / NC, c = blockIdx.x % NC, d = threadIdx.x;
  const float dec = decay[d];
  float f = dec;
  const float mi = cin[((size_t)b * NC + c) * Dd + d];
  const size_t base = ((size_t)b * Tt + (size_t)c * Lc) * Dd + d;
  for (int i = 0; i < Lc; ++i) {
    const size_t idx = base + (size_t)i * Dd;
    const float mem = kvloc[idx] + f * mi;
    f *= dec;
    out[idx] = x[idx] + (float)q16[idx] * mem;
  }
}

// ---------------------------------------------------------------------------
// K5: fused LN2 + MLP layer 1 (D->H) with ReLU, hidden stored f16.
// ---------------------------------------------------------------------------
__global__ __launch_bounds__(256) void k_ln_mlp1(
    const float* __restrict__ xr, const float* __restrict__ g,
    const float* __restrict__ be, const half_t* __restrict__ W1,
    const float* __restrict__ b1, half_t* __restrict__ hid) {
  __shared__ half_t hA[32 * HSTR];
  const int tok0 = blockIdx.x * 32;
  const int wave = threadIdx.x >> 5, lane = threadIdx.x & 31;

#pragma unroll
  for (int rr = 0; rr < 4; ++rr) {
    const int row = wave * 4 + rr;
    const float* p = xr + (size_t)(tok0 + row) * Dd;
    float xv[8], s = 0.f, sq = 0.f;
#pragma unroll
    for (int i = 0; i < 8; ++i) {
      float v = p[lane + 32 * i];
      xv[i] = v; s += v; sq += v * v;
    }
#pragma unroll
    for (int o = 16; o > 0; o >>= 1) {
      s += __shfl_xor(s, o, 32);
      sq += __shfl_xor(sq, o, 32);
    }
    const float mu = s * (1.f / Dd);
    const float rstd = rsqrtf(sq * (1.f / Dd) - mu * mu + 1e-5f);
#pragma unroll
    for (int i = 0; i < 8; ++i) {
      const int c = lane + 32 * i;
      hA[row * HSTR + c] = (half_t)((xv[i] - mu) * rstd * g[c] + be[c]);
    }
  }
  __syncthreads();

  for (int mt = 0; mt < 2; ++mt) {
    for (int ni = 0; ni < 8; ++ni) {
      const int nt = wave * 8 + ni;
      __builtin_prefetch(W1 + (size_t)(nt * 16 + (lane & 15)) * Dd, 0, 1);
      v8f acc = {};
#pragma unroll
      for (int kk = 0; kk < 8; ++kk) {
        const v16h af = frag_rows(hA, HSTR, mt * 16, lane, kk * 32);
        const v16h bf = frag_rows(W1, Dd, nt * 16, lane, kk * 32);
        acc = wmma_f16(af, bf, acc);
      }
      const int col = nt * 16 + (lane & 15);
      const float bb = b1[col];
#pragma unroll
      for (int r = 0; r < 8; ++r) {
        const int row = tok0 + mt * 16 + r + ((lane & 16) ? 8 : 0);
        const float v = acc[r] + bb;
        hid[(size_t)row * Hh + col] = (half_t)(v > 0.f ? v : 0.f);
      }
    }
  }
}

// ---------------------------------------------------------------------------
// K6: MLP layer 2 (H->D) + residual.  32x1024 f16 hidden tile staged in LDS
// by the Tensor Data Mover (with LDS padding for bank-conflict-free A reads).
// ---------------------------------------------------------------------------
__global__ __launch_bounds__(256) void k_mlp2_res(const half_t* __restrict__ hid,
                                                  const half_t* __restrict__ W2,
                                                  const float* __restrict__ b2,
                                                  float* __restrict__ out) {
  __shared__ uint32_t shH[16704];   // 32*512 dwords + 5 pad dwords / 256 dwords
  const int tok0 = blockIdx.x * 32;
  const int wave = threadIdx.x >> 5, lane = threadIdx.x & 31;

#if USE_TDM
  if (wave == 0) {   // TDM is wave-level DMA (EXEC-independent): one wave issues
    const uint32_t lds = (uint32_t)(uintptr_t)(&shH[0]);
    const uint64_t ga = (uint64_t)(uintptr_t)(hid + (size_t)tok0 * Hh);
    u32x4 g0;
    g0[0] = 1u;                                           // count=1, user desc
    g0[1] = lds;                                          // lds_addr
    g0[2] = (uint32_t)ga;                                 // global_addr[31:0]
    g0[3] = (uint32_t)((ga >> 32) & 0x1FFFFFFu) | (2u << 30); // addr hi | type=2
    i32x8 g1;
    // data_size=2B, pad_enable, pad_interval=256dw, pad_amount=5dw
    g1[0] = (int)((1u << 16) | (1u << 20) | (7u << 22) | (4u << 25));
    g1[1] = (int)((uint32_t)Hh << 16);                    // tensor_dim0 lo16
    g1[2] = 0;                                            // dim0 hi / dim1 lo16 (=0)
    g1[3] = (int)(((uint32_t)Hh << 16) | 1u);             // dim1 hi16=1 | tile_dim0=1024
    g1[4] = 32;                                           // tile_dim1 = 32 rows
    g1[5] = Hh;                                           // tensor_dim0_stride
    g1[6] = 0;
    g1[7] = 0;
    const i32x4 z4 = {0, 0, 0, 0};                        // groups 2/3 disabled (2D)
    const i32x8 z8 = {0, 0, 0, 0, 0, 0, 0, 0};           // unused final group
    __builtin_amdgcn_tensor_load_to_lds(g0, g1, z4, z4, z8, 0);
    __builtin_amdgcn_s_wait_tensorcnt((short)0);
  }
  __syncthreads();
#else
  const uint32_t* src = (const uint32_t*)(hid + (size_t)tok0 * Hh);
  for (int idx = threadIdx.x; idx < 32 * 512; idx += 256)
    shH[idx + 5 * (idx >> 8)] = src[idx];
  __syncthreads();
#endif

  for (int mt = 0; mt < 2; ++mt) {
    for (int ni = 0; ni < 2; ++ni) {
      const int nt = wave * 2 + ni;
      __builtin_prefetch(W2 + (size_t)(nt * 16 + (lane & 15)) * Hh, 0, 1);
      v8f acc = {};
      for (int kk = 0; kk < 32; ++kk) {
        const v16h af = frag_padded(shH, mt * 16, lane, kk * 32);
        const v16h bf = frag_rows(W2, Hh, nt * 16, lane, kk * 32);
        acc = wmma_f16(af, bf, acc);
      }
      const int col = nt * 16 + (lane & 15);
      const float bb = b2[col];
#pragma unroll
      for (int r = 0; r < 8; ++r) {
        const int row = tok0 + mt * 16 + r + ((lane & 16) ? 8 : 0);
        const size_t idx = (size_t)row * Dd + col;
        out[idx] = out[idx] + acc[r] + bb;   // residual: xr + mlp
      }
    }
  }
}

// ---------------------------------------------------------------------------
extern "C" void kernel_launch(void* const* d_in, const int* in_sizes, int n_in,
                              void* d_out, int out_size, void* d_ws, size_t ws_size,
                              hipStream_t stream) {
  (void)in_sizes; (void)n_in; (void)out_size; (void)ws_size;
  const float* x     = (const float*)d_in[0];
  const float* Wq    = (const float*)d_in[1];
  const float* bq    = (const float*)d_in[2];
  const float* Wk    = (const float*)d_in[3];
  const float* bk    = (const float*)d_in[4];
  const float* Wv    = (const float*)d_in[5];
  const float* bv    = (const float*)d_in[6];
  const float* decay = (const float*)d_in[7];
  const float* ln1g  = (const float*)d_in[8];
  const float* ln1b  = (const float*)d_in[9];
  const float* W1    = (const float*)d_in[10];
  const float* b1    = (const float*)d_in[11];
  const float* W2    = (const float*)d_in[12];
  const float* b2    = (const float*)d_in[13];
  const float* ln2g  = (const float*)d_in[14];
  const float* ln2b  = (const float*)d_in[15];

  char* ws = (char*)d_ws;
  half_t* Wq16 = (half_t*)(ws + OFF_WQ);
  half_t* Wk16 = (half_t*)(ws + OFF_WK);
  half_t* Wv16 = (half_t*)(ws + OFF_WV);
  half_t* W116 = (half_t*)(ws + OFF_W1);
  half_t* W216 = (half_t*)(ws + OFF_W2);
  half_t* q16  = (half_t*)(ws + OFF_Q16);
  float*  kv   = (float*)(ws + OFF_KV);
  float*  car  = (float*)(ws + OFF_CAR);
  float*  cin  = (float*)(ws + OFF_CIN);
  half_t* hid  = (half_t*)(ws + OFF_HID);
  float*  out  = (float*)d_out;

  k_cvt_f16<<<(Dd * Dd + 255) / 256, 256, 0, stream>>>(Wq, Wq16, Dd * Dd);
  k_cvt_f16<<<(Dd * Dd + 255) / 256, 256, 0, stream>>>(Wk, Wk16, Dd * Dd);
  k_cvt_f16<<<(Dd * Dd + 255) / 256, 256, 0, stream>>>(Wv, Wv16, Dd * Dd);
  k_cvt_f16<<<(Hh * Dd + 255) / 256, 256, 0, stream>>>(W1, W116, Hh * Dd);
  k_cvt_f16<<<(Dd * Hh + 255) / 256, 256, 0, stream>>>(W2, W216, Dd * Hh);

  k_ln_qkv<<<BT / 32, 256, 0, stream>>>(x, ln1g, ln1b, Wq16, Wk16, Wv16,
                                        bq, bk, bv, q16, kv);
  k_scan_chunks<<<Bb * NC, 256, 0, stream>>>(kv, decay, car);
  k_scan_carry<<<Bb, 256, 0, stream>>>(car, cin, decay);
  k_ret_residual<<<Bb * NC, 256, 0, stream>>>(x, q16, kv, cin, decay, out);
  k_ln_mlp1<<<BT / 32, 256, 0, stream>>>(out, ln2g, ln2b, W116, b1, hid);
  k_mlp2_res<<<BT / 32, 256, 0, stream>>>(hid, W216, b2, out);
}
